// Multi_head_attention_51015621542503
// MI455X (gfx1250) — matrix operand order repaired and checked by reading, not compiled, not run on hardware
//
#include <hip/hip_runtime.h>
#include <hip/hip_bf16.h>

// ---------------------------------------------------------------------------
// Multi-head attention for MI455X (gfx1250), bf16 WMMA pipeline:
//   1) convert fp32 -> bf16 (X, Wq, Wk, Wv, Wo)
//   2) Q/K/V projection GEMMs (v_wmma_f32_16x16x32_bf16), TDM-fed, double-buffered
//   3) flash attention (causal, online softmax), TDM K-tiles, ctx -> [b,s,d] bf16
//   4) out = ctx @ Wo + bo (fp32)
// ---------------------------------------------------------------------------

typedef __attribute__((ext_vector_type(16))) __bf16 v16bf;
typedef __attribute__((ext_vector_type(8)))  float  v8f;
typedef __attribute__((ext_vector_type(4)))  unsigned int u32x4;
typedef __attribute__((ext_vector_type(8)))  int    i32x8;
typedef __attribute__((ext_vector_type(4)))  int    i32x4;

union Frag16 {           // 16 bf16 elements == 8 VGPRs
    v16bf        v;
    unsigned int u[8];
};

#define B_     4
#define S_     2048
#define D_     1024
#define H_     16
#define HD_    64
#define M_     (B_ * S_)            // 8192 rows

__device__ __forceinline__ unsigned short f2bf(float f) {
    unsigned int u = __float_as_uint(f);
    unsigned int r = u + 0x7FFFu + ((u >> 16) & 1u);   // round-to-nearest-even
    return (unsigned short)(r >> 16);
}

// ---------------------------------------------------------------------------
// TDM: 2D tile load, global(bf16) -> LDS, per CDNA5 D# spec (08_async_tensor
// §8.3/8.4). tileW elements per row (tileW*2 bytes, multiple of 4B), tileH
// rows, row stride strideElems elements. Tensor dims == tile dims (no OOB).
// clang-23 toolchain: 6-arg builtin (g0 u32x4, g1 i32x8, g2 i32x4, g3 i32x4,
// extra i32x8 zero-filled, cpol i32).
// ---------------------------------------------------------------------------
#if defined(__has_builtin)
#if __has_builtin(__builtin_amdgcn_tensor_load_to_lds) && __has_builtin(__builtin_amdgcn_s_wait_tensorcnt)
#define USE_TDM 1
#endif
#endif

#ifdef USE_TDM
__device__ __forceinline__ void tdm_load_2d(void* ldsPtr, const void* gPtr,
                                            unsigned int tileW, unsigned int tileH,
                                            unsigned int strideElems) {
    unsigned long long ga = (unsigned long long)(size_t)gPtr;
    unsigned int ldsOff = (unsigned int)(size_t)ldsPtr;   // low 32b of flat = LDS addr
    u32x4 g0;
    g0.x = 1u;                                            // count=1, user desc
    g0.y = ldsOff;                                        // [63:32] lds_addr
    g0.z = (unsigned int)ga;                              // [95:64] global_addr lo
    g0.w = ((unsigned int)(ga >> 32) & 0x01FFFFFFu) | (2u << 30); // addr hi | type=2
    i32x8 g1;
    g1[0] = (int)(1u << 16);                              // data_size=1 (2 bytes)
    g1[1] = (int)(tileW << 16);                           // tensor_dim0 lo16
    g1[2] = (int)((tileW >> 16) | (tileH << 16));         // tensor_dim0 hi | dim1 lo
    g1[3] = (int)((tileH >> 16) | (tileW << 16));         // dim1 hi | tile_dim0
    g1[4] = (int)tileH;                                   // tile_dim1 (tile_dim2=0)
    g1[5] = (int)strideElems;                             // tensor_dim0_stride lo32
    g1[6] = 0;
    g1[7] = 0;
    i32x4 z4 = {0, 0, 0, 0};                              // groups 2/3 unused (2D)
    i32x8 z8 = {0, 0, 0, 0, 0, 0, 0, 0};
    __builtin_amdgcn_tensor_load_to_lds(g0, g1, z4, z4, z8, 0);
}
#define TDM_WAIT() __builtin_amdgcn_s_wait_tensorcnt(0)
#endif

// ---------------------------------------------------------------------------
// Kernel 1: fp32 -> bf16 conversion
// ---------------------------------------------------------------------------
__global__ __launch_bounds__(256) void cvt_f32_bf16_kernel(
    const float* __restrict__ src, unsigned short* __restrict__ dst, int n) {
    int i = blockIdx.x * 256 + threadIdx.x;
    if (i < n) dst[i] = f2bf(src[i]);
}

// ---------------------------------------------------------------------------
// GEMM tiling: block tile 128(M) x 64(N), K-step 32, 256 threads = 8 waves,
// each wave owns a 32x32 tile = 2x2 WMMA accumulators. A tile via TDM,
// B tile manual transpose-scatter, both double-buffered in LDS.
// ---------------------------------------------------------------------------
#define GM_BM 128
#define GM_BN 64
#define GM_BK 32

__device__ __forceinline__ void gemm_load_B(const unsigned short* __restrict__ W,
                                            unsigned short* Bs, int n0, int k0,
                                            int tid) {
    int kk = tid >> 3, nn0 = (tid & 7) * 8;
    const unsigned short* src = W + (size_t)(k0 + kk) * D_ + n0 + nn0;
#pragma unroll
    for (int j = 0; j < 8; ++j) Bs[(nn0 + j) * GM_BK + kk] = src[j];
}

#ifndef USE_TDM
__device__ __forceinline__ void gemm_load_A_fallback(
    const unsigned short* __restrict__ A, unsigned short* As, int m0, int k0,
    int tid) {
    int row = tid >> 1, c0 = (tid & 1) * 16;
    const uint4* src =
        reinterpret_cast<const uint4*>(A + (size_t)(m0 + row) * D_ + k0 + c0);
    uint4* d = reinterpret_cast<uint4*>(As + row * GM_BK + c0);
    d[0] = src[0];
    d[1] = src[1];
}
#endif

template <bool WITH_BIAS>
__device__ __forceinline__ void gemm_body(
    const unsigned short* __restrict__ A,    // [8192,1024] bf16
    const unsigned short* __restrict__ W,    // [1024,1024] bf16
    const float* __restrict__ bias,          // [1024] or nullptr
    unsigned short* __restrict__ dstBf,      // proj output [b,h,s,hd]
    float* __restrict__ dstF) {              // out output [8192,1024]
    __shared__ __align__(16) unsigned short As[2][GM_BM * GM_BK]; // [m][k]
    __shared__ __align__(16) unsigned short Bs[2][GM_BN * GM_BK]; // [n][k]

    const int tid  = threadIdx.x;
    const int lane = tid & 31;
    const int wave = tid >> 5;
    const int m0   = blockIdx.x * GM_BM;
    const int n0   = blockIdx.y * GM_BN;
    const int wm   = (wave >> 1) * 32;
    const int wn   = (wave & 1) * 32;
    const int lm   = lane & 15;
    const int lh   = lane >> 4;

    v8f acc[2][2];
#pragma unroll
    for (int i = 0; i < 2; ++i)
#pragma unroll
        for (int j = 0; j < 2; ++j) acc[i][j] = (v8f)0.0f;

    // prologue: stage buffer 0
#ifdef USE_TDM
    if (wave == 0)
        tdm_load_2d(&As[0][0], A + (size_t)m0 * D_, GM_BK, GM_BM, D_);
#else
    gemm_load_A_fallback(A, &As[0][0], m0, 0, tid);
#endif
    gemm_load_B(W, &Bs[0][0], n0, 0, tid);
#ifdef USE_TDM
    if (wave == 0) TDM_WAIT();
#endif
    __syncthreads();

    const int NIT = D_ / GM_BK;  // 32
    for (int it = 0; it < NIT; ++it) {
        const int cur = it & 1, nxt = cur ^ 1;
        if (it + 1 < NIT) {      // prefetch next tiles into alternate buffers
            const int k0n = (it + 1) * GM_BK;
#ifdef USE_TDM
            if (wave == 0)
                tdm_load_2d(&As[nxt][0], A + (size_t)m0 * D_ + k0n, GM_BK, GM_BM, D_);
#else
            gemm_load_A_fallback(A, &As[nxt][0], m0, k0n, tid);
#endif
            gemm_load_B(W, &Bs[nxt][0], n0, k0n, tid);
        }

        Frag16 af[2], bf[2];
#pragma unroll
        for (int i = 0; i < 2; ++i)
#pragma unroll
            for (int v = 0; v < 8; ++v) {
                int kk = ((v & 4) << 2) + (lh << 3) + ((v & 3) << 1);
                af[i].u[v] = *reinterpret_cast<const unsigned int*>(
                    &As[cur][(wm + i * 16 + lm) * GM_BK + kk]);
            }
#pragma unroll
        for (int j = 0; j < 2; ++j)
#pragma unroll
            for (int v = 0; v < 8; ++v) {
                int kk = ((v & 4) << 2) + (lh << 3) + ((v & 3) << 1);
                bf[j].u[v] = *reinterpret_cast<const unsigned int*>(
                    &Bs[cur][(wn + j * 16 + lm) * GM_BK + kk]);
            }
#pragma unroll
        for (int i = 0; i < 2; ++i)
#pragma unroll
            for (int j = 0; j < 2; ++j)
                acc[i][j] = __builtin_amdgcn_wmma_f32_16x16x32_bf16(
                    false, af[i].v, false, bf[j].v, (short)0, acc[i][j], false, false);

#ifdef USE_TDM
        if (wave == 0 && it + 1 < NIT) TDM_WAIT();
#endif
        __syncthreads();
    }

    // epilogue
#pragma unroll
    for (int i = 0; i < 2; ++i)
#pragma unroll
        for (int j = 0; j < 2; ++j)
#pragma unroll
            for (int r = 0; r < 8; ++r) {
                int m = m0 + wm + i * 16 + r + 8 * lh;
                int n = n0 + wn + j * 16 + lm;
                if (WITH_BIAS) {
                    dstF[(size_t)m * D_ + n] = acc[i][j][r] + bias[n];
                } else {
                    int bb = m >> 11, srow = m & (S_ - 1);
                    int hh = n >> 6,  hd   = n & (HD_ - 1);
                    dstBf[((size_t)((bb * H_ + hh) * S_ + srow)) * HD_ + hd] =
                        f2bf(acc[i][j][r]);
                }
            }
}

// Kernel 2: projection GEMM, output bf16 [b,h,s,hd]
__global__ __launch_bounds__(256) void gemm_proj_kernel(
    const unsigned short* __restrict__ A, const unsigned short* __restrict__ W,
    unsigned short* __restrict__ dst) {
    gemm_body<false>(A, W, nullptr, dst, nullptr);
}

// Kernel 4: out = ctx @ Wo + bo, fp32
__global__ __launch_bounds__(256) void gemm_out_kernel(
    const unsigned short* __restrict__ A, const unsigned short* __restrict__ W,
    const float* __restrict__ bias, float* __restrict__ out) {
    gemm_body<true>(A, W, bias, nullptr, out);
}

// ---------------------------------------------------------------------------
// Kernel 3: flash attention, causal.  grid (32 qtiles, 16 h, 4 b), 128 thr.
// Each wave handles 16 q rows; key blocks of 32. K tile via TDM.
// ---------------------------------------------------------------------------
__global__ __launch_bounds__(128) void flash_attn_kernel(
    const unsigned short* __restrict__ Qb,   // [b,h,s,hd] bf16
    const unsigned short* __restrict__ Kb,
    const unsigned short* __restrict__ Vb,
    unsigned short* __restrict__ Ctx) {      // [b,s,d] bf16
    __shared__ __align__(16) unsigned short Kt[32 * 64];      // [key][hd]
    __shared__ __align__(16) unsigned short Vt[64 * 32];      // [hd][key]
    __shared__ __align__(16) unsigned short Ps[4 * 16 * 32];  // per-wave P tile

    const int tid  = threadIdx.x;
    const int lane = tid & 31;
    const int wave = tid >> 5;
    const int qt   = blockIdx.x;
    const int h    = blockIdx.y;
    const int b    = blockIdx.z;
    const int lm   = lane & 15;
    const int lh   = lane >> 4;
    const size_t headBase = ((size_t)(b * H_ + h)) * S_ * HD_;
    const int qrow0 = qt * 64 + wave * 16;

    // Q fragments: 2 hd-slices of 32
    Frag16 aq[2];
#pragma unroll
    for (int s = 0; s < 2; ++s)
#pragma unroll
        for (int v = 0; v < 8; ++v) {
            int kk = s * 32 + ((v & 4) << 2) + (lh << 3) + ((v & 3) << 1);
            aq[s].u[v] = *reinterpret_cast<const unsigned int*>(
                Qb + headBase + (size_t)(qrow0 + lm) * HD_ + kk);
        }

    v8f ctx[4];
#pragma unroll
    for (int c = 0; c < 4; ++c) ctx[c] = (v8f)0.0f;
    float rowMax[8], rowSum[8];
#pragma unroll
    for (int r = 0; r < 8; ++r) { rowMax[r] = -1e30f; rowSum[r] = 0.0f; }

    const int nblocks = 2 * (qt + 1);
    for (int kb = 0; kb < nblocks; ++kb) {
        const int k0 = kb * 32;
        // K block: TDM natural layout; V block: manual transposed scatter
#ifdef USE_TDM
        if (wave == 0)
            tdm_load_2d(&Kt[0], Kb + headBase + (size_t)k0 * HD_, HD_, 32, HD_);
#endif
        {
            int key = tid >> 2, c0 = (tid & 3) * 16;
#ifndef USE_TDM
            const uint4* ks = reinterpret_cast<const uint4*>(
                Kb + headBase + (size_t)(k0 + key) * HD_ + c0);
            uint4* kd = reinterpret_cast<uint4*>(Kt + key * 64 + c0);
            kd[0] = ks[0];
            kd[1] = ks[1];
#endif
            const unsigned short* vs =
                Vb + headBase + (size_t)(k0 + key) * HD_ + c0;
#pragma unroll
            for (int j = 0; j < 16; ++j) Vt[(c0 + j) * 32 + key] = vs[j];
        }
#ifdef USE_TDM
        if (wave == 0) TDM_WAIT();
#endif
        __syncthreads();

        // S = Q K^T  (two 16-key subtiles)
        float pv[2][8];
#pragma unroll
        for (int sub = 0; sub < 2; ++sub) {
            v8f s = (v8f)0.0f;
#pragma unroll
            for (int sl = 0; sl < 2; ++sl) {
                Frag16 bk;
#pragma unroll
                for (int v = 0; v < 8; ++v) {
                    int kk = sl * 32 + ((v & 4) << 2) + (lh << 3) + ((v & 3) << 1);
                    bk.u[v] = *reinterpret_cast<const unsigned int*>(
                        Kt + (sub * 16 + lm) * 64 + kk);
                }
                s = __builtin_amdgcn_wmma_f32_16x16x32_bf16(
                    false, aq[sl].v, false, bk.v, (short)0, s, false, false);
            }
            // causal mask + 1/sqrt(hd) scale (C layout: m = r + 8*lh, n = lm)
#pragma unroll
            for (int r = 0; r < 8; ++r) {
                int mg = qrow0 + r + 8 * lh;
                int ng = k0 + sub * 16 + lm;
                pv[sub][r] = (ng <= mg) ? s[r] * 0.125f : -1e30f;
            }
        }

        // online softmax: row max / rescale / row sum
        float alpha[8];
#pragma unroll
        for (int r = 0; r < 8; ++r) {
            float t = fmaxf(pv[0][r], pv[1][r]);
            t = fmaxf(t, __shfl_xor(t, 1));
            t = fmaxf(t, __shfl_xor(t, 2));
            t = fmaxf(t, __shfl_xor(t, 4));
            t = fmaxf(t, __shfl_xor(t, 8));
            float nm = fmaxf(rowMax[r], t);
            alpha[r] = __expf(rowMax[r] - nm);
            rowMax[r] = nm;
            pv[0][r] = __expf(pv[0][r] - nm);
            pv[1][r] = __expf(pv[1][r] - nm);
            float ssum = pv[0][r] + pv[1][r];
            ssum += __shfl_xor(ssum, 1);
            ssum += __shfl_xor(ssum, 2);
            ssum += __shfl_xor(ssum, 4);
            ssum += __shfl_xor(ssum, 8);
            rowSum[r] = rowSum[r] * alpha[r] + ssum;
        }

        // P (C layout) -> LDS -> A-fragment layout
#pragma unroll
        for (int sub = 0; sub < 2; ++sub)
#pragma unroll
            for (int r = 0; r < 8; ++r)
                Ps[wave * 512 + (r + 8 * lh) * 32 + sub * 16 + lm] =
                    f2bf(pv[sub][r]);
        __syncthreads();

        Frag16 ap;
#pragma unroll
        for (int v = 0; v < 8; ++v) {
            int kk = ((v & 4) << 2) + (lh << 3) + ((v & 3) << 1);
            ap.u[v] = *reinterpret_cast<const unsigned int*>(
                Ps + wave * 512 + lm * 32 + kk);
        }

        // ctx = ctx*alpha + P @ V  (4 hd-chunks of 16)
#pragma unroll
        for (int c = 0; c < 4; ++c) {
#pragma unroll
            for (int r = 0; r < 8; ++r) ctx[c][r] *= alpha[r];
            Frag16 bv;
#pragma unroll
            for (int v = 0; v < 8; ++v) {
                int kk = ((v & 4) << 2) + (lh << 3) + ((v & 3) << 1);
                bv.u[v] = *reinterpret_cast<const unsigned int*>(
                    Vt + (c * 16 + lm) * 32 + kk);
            }
            ctx[c] = __builtin_amdgcn_wmma_f32_16x16x32_bf16(
                false, ap.v, false, bv.v, (short)0, ctx[c], false, false);
        }
        __syncthreads();
    }

    // epilogue: normalize, write ctx to [b, s, d] (heads re-merged)
#pragma unroll
    for (int c = 0; c < 4; ++c)
#pragma unroll
        for (int r = 0; r < 8; ++r) {
            float o = ctx[c][r] / rowSum[r];
            int q = qrow0 + r + 8 * lh;
            int n = c * 16 + lm;
            Ctx[((size_t)(b * S_ + q)) * D_ + h * HD_ + n] = f2bf(o);
        }
}

// ---------------------------------------------------------------------------
// Launch
// ---------------------------------------------------------------------------
extern "C" void kernel_launch(void* const* d_in, const int* in_sizes, int n_in,
                              void* d_out, int out_size, void* d_ws, size_t ws_size,
                              hipStream_t stream) {
    const float* X  = (const float*)d_in[0];   // [4,2048,1024]
    const float* Wq = (const float*)d_in[1];   // [1024,1024]
    const float* Wk = (const float*)d_in[2];
    const float* Wv = (const float*)d_in[3];
    const float* Wo = (const float*)d_in[4];
    const float* bo = (const float*)d_in[5];   // [1024]
    float* out = (float*)d_out;                // [4,2048,1024]

    const size_t nX = (size_t)M_ * D_;         // 8,388,608
    const size_t nW = (size_t)D_ * D_;         // 1,048,576

    unsigned short* Xb   = (unsigned short*)d_ws;
    unsigned short* Wqb  = Xb  + nX;
    unsigned short* Wkb  = Wqb + nW;
    unsigned short* Wvb  = Wkb + nW;
    unsigned short* Wob  = Wvb + nW;
    unsigned short* Qb   = Wob + nW;
    unsigned short* Kb   = Qb  + nX;
    unsigned short* Vb   = Kb  + nX;
    unsigned short* Ctxb = Vb  + nX;
    (void)ws_size; (void)in_sizes; (void)n_in; (void)out_size;

    // 1) conversions
    cvt_f32_bf16_kernel<<<(int)(nX / 256), 256, 0, stream>>>(X,  Xb,  (int)nX);
    cvt_f32_bf16_kernel<<<(int)(nW / 256), 256, 0, stream>>>(Wq, Wqb, (int)nW);
    cvt_f32_bf16_kernel<<<(int)(nW / 256), 256, 0, stream>>>(Wk, Wkb, (int)nW);
    cvt_f32_bf16_kernel<<<(int)(nW / 256), 256, 0, stream>>>(Wv, Wvb, (int)nW);
    cvt_f32_bf16_kernel<<<(int)(nW / 256), 256, 0, stream>>>(Wo, Wob, (int)nW);

    // 2) Q/K/V projections
    dim3 gGemm(M_ / GM_BM, D_ / GM_BN);        // (64, 16)
    gemm_proj_kernel<<<gGemm, 256, 0, stream>>>(Xb, Wqb, Qb);
    gemm_proj_kernel<<<gGemm, 256, 0, stream>>>(Xb, Wkb, Kb);
    gemm_proj_kernel<<<gGemm, 256, 0, stream>>>(Xb, Wvb, Vb);

    // 3) flash attention
    dim3 gAttn(S_ / 64, H_, B_);               // (32, 16, 4)
    flash_attn_kernel<<<gAttn, 128, 0, stream>>>(Qb, Kb, Vb, Ctxb);

    // 4) output projection + bias
    gemm_out_kernel<<<gGemm, 256, 0, stream>>>(Ctxb, Wob, bo, out);
}
